// Decoder_33277406609768
// MI455X (gfx1250) — compile-verified, matrix operand order checked by reading, not run on hardware
//
#include <hip/hip_runtime.h>
#include <hip/hip_bf16.h>
#include <cmath>

#define L_  6
#define D_  768
#define DI_ 1024
#define H_  8
#define HD_ 96
#define FF_ 3072
#define V_  32000
#define B_  8
#define S_  512
#define P_  196

typedef __bf16 bf16x16 __attribute__((ext_vector_type(16)));
typedef float f32x8 __attribute__((ext_vector_type(8)));
typedef float f32x4 __attribute__((ext_vector_type(4)));

__device__ __forceinline__ unsigned short f2bf(float f) {
    unsigned int u = __float_as_uint(f);
    unsigned int r = u + 0x7FFFu + ((u >> 16) & 1u);   // round-to-nearest-even
    return (unsigned short)(r >> 16);
}
__device__ __forceinline__ unsigned int pack2(float a, float b) {
    return (unsigned int)f2bf(a) | ((unsigned int)f2bf(b) << 16);
}

// Fragment loader for the CDNA5 16-bit A-matrix 16x32 layout:
//   lanes 0-15 : row = lo, K = {0..7} then {16..23}
//   lanes 16-31: row = lo, K = {8..15} then {24..31}
// LDS tile is [row][K] with row stride 40 halves (80B, 16B-aligned rows).
__device__ __forceinline__ bf16x16 load_frag(const unsigned short* smem,
                                             int row0, int lo, int hi) {
    const uint4* p = (const uint4*)(smem + (row0 + lo) * 40 + hi * 8);
    union { uint4 q[2]; bf16x16 v; } u;
    u.q[0] = p[0];   // K = hi*8 .. hi*8+7
    u.q[1] = p[2];   // K = 16+hi*8 .. 16+hi*8+7
    return u.v;
}

// Generic batched GEMM: C = A(MxK) * B(KxN) [+ bias] [relu], bf16 WMMA, f32 acc.
// Workgroup tile: 128(M) x 256(N); 8 waves, each wave 64x64 = 4x4 WMMA tiles.
// A: row-major, element (m,k) at A[m*ldA + k], batch base = go*sAo + gi*sAi.
// B: element (k,n) at B[k*ldbK + n*ldbN]  (covers [K,N] weights and [N,K] K-matrices).
// ALIGNED=true  : M%128==0, N%256==0, K%32==0 -> zero guards anywhere.
// ALIGNED=false : fast path for interior tiles, branchless clamped loads on edges.
template <bool ALIGNED>
__global__ __launch_bounds__(256) void wmma_gemm(
    const float* __restrict__ A, long ldA, long sAo, long sAi,
    const float* __restrict__ Bg, long ldbK, long ldbN, long sBo, long sBi,
    const float* __restrict__ bias,
    float* __restrict__ C, long ldC, long sCo, long sCi,
    int M, int N, int K, int inner, int relu)
{
    __shared__ unsigned short As[128 * 40];
    __shared__ unsigned short Bs[256 * 40];

    const int g  = blockIdx.z;
    const int go = g / inner, gi = g % inner;
    const float* Ab = A  + (long)go * sAo + (long)gi * sAi;
    const float* Bb = Bg + (long)go * sBo + (long)gi * sBi;
    float*       Cb = C  + (long)go * sCo + (long)gi * sCi;

    const int m0 = blockIdx.y * 128;
    const int n0 = blockIdx.x * 256;
    const bool mnfull = ALIGNED || ((m0 + 128 <= M) && (n0 + 256 <= N));

    const int tid  = threadIdx.x;
    const int r    = tid >> 1;          // 0..127 (row of tile being staged)
    const int kk0  = (tid & 1) * 16;    // which 16-wide K half

    const int wave = tid >> 5;
    const int wm   = wave >> 2;         // 0..1 : 64-row slab
    const int wn   = wave & 3;          // 0..3 : 64-col slab
    const int lane = tid & 31;
    const int lo   = lane & 15;
    const int hi   = lane >> 4;

    f32x8 acc[4][4] = {};

    for (int k0 = 0; k0 < K; k0 += 32) {
        unsigned int* da = (unsigned int*)(As + r * 40 + kk0);
        const bool full = ALIGNED || (mnfull && (k0 + 32 <= K));  // block-uniform

        if (full) {
            // ---- fast path: unpredicated vector loads ----
            const float* ap = Ab + (long)(m0 + r) * ldA + (k0 + kk0);
            f32x4 a0 = ((const f32x4*)ap)[0];
            f32x4 a1 = ((const f32x4*)ap)[1];
            f32x4 a2 = ((const f32x4*)ap)[2];
            f32x4 a3 = ((const f32x4*)ap)[3];
            if (k0 + 32 < K) __builtin_prefetch(ap + 32, 0, 0);
            da[0] = pack2(a0[0], a0[1]); da[1] = pack2(a0[2], a0[3]);
            da[2] = pack2(a1[0], a1[1]); da[3] = pack2(a1[2], a1[3]);
            da[4] = pack2(a2[0], a2[1]); da[5] = pack2(a2[2], a2[3]);
            da[6] = pack2(a3[0], a3[1]); da[7] = pack2(a3[2], a3[3]);

            #pragma unroll
            for (int hB = 0; hB < 2; ++hB) {
                const int n = r + hB * 128;
                unsigned int* db = (unsigned int*)(Bs + n * 40 + kk0);
                if (ldbK == 1) {
                    // B stored [N,K]: 16 contiguous floats per thread
                    const float* bp = Bb + (long)(n0 + n) * ldbN + (k0 + kk0);
                    f32x4 b0 = ((const f32x4*)bp)[0];
                    f32x4 b1 = ((const f32x4*)bp)[1];
                    f32x4 b2 = ((const f32x4*)bp)[2];
                    f32x4 b3 = ((const f32x4*)bp)[3];
                    db[0] = pack2(b0[0], b0[1]); db[1] = pack2(b0[2], b0[3]);
                    db[2] = pack2(b1[0], b1[1]); db[3] = pack2(b1[2], b1[3]);
                    db[4] = pack2(b2[0], b2[1]); db[5] = pack2(b2[2], b2[3]);
                    db[6] = pack2(b3[0], b3[1]); db[7] = pack2(b3[2], b3[3]);
                } else {
                    // B stored [K,N] (ldbN==1 for weights): coalesced across threads
                    const float* bp = Bb + (long)(k0 + kk0) * ldbK + (n0 + n);
                    unsigned short tmp[16];
                    #pragma unroll
                    for (int i = 0; i < 16; ++i) tmp[i] = f2bf(bp[(long)i * ldbK]);
                    #pragma unroll
                    for (int j = 0; j < 8; ++j)
                        db[j] = (unsigned int)tmp[2 * j] | ((unsigned int)tmp[2 * j + 1] << 16);
                }
            }
        } else if (!ALIGNED) {
            // ---- edge path: branchless clamped loads + select-zero ----
            unsigned short tmp[16];
            const int arow   = m0 + r;
            const int arow_c = min(arow, M - 1);
            const float* ap  = Ab + (long)arow_c * ldA;
            #pragma unroll
            for (int i = 0; i < 16; ++i) {
                const int kk = k0 + kk0 + i;
                const int kc = min(kk, K - 1);
                float v = ap[kc];
                v = (arow < M && kk < K) ? v : 0.f;
                tmp[i] = f2bf(v);
            }
            #pragma unroll
            for (int j = 0; j < 8; ++j)
                da[j] = (unsigned int)tmp[2 * j] | ((unsigned int)tmp[2 * j + 1] << 16);

            #pragma unroll
            for (int hB = 0; hB < 2; ++hB) {
                const int n    = r + hB * 128;
                const int bn   = n0 + n;
                const int bn_c = min(bn, N - 1);
                const float* bp = Bb + (long)bn_c * ldbN;
                unsigned int* db = (unsigned int*)(Bs + n * 40 + kk0);
                #pragma unroll
                for (int i = 0; i < 16; ++i) {
                    const int kk = k0 + kk0 + i;
                    const int kc = min(kk, K - 1);
                    float v = bp[(long)kc * ldbK];
                    v = (bn < N && kk < K) ? v : 0.f;
                    tmp[i] = f2bf(v);
                }
                #pragma unroll
                for (int j = 0; j < 8; ++j)
                    db[j] = (unsigned int)tmp[2 * j] | ((unsigned int)tmp[2 * j + 1] << 16);
            }
        }
        __syncthreads();

        bf16x16 af[4], bfv[4];
        #pragma unroll
        for (int tm = 0; tm < 4; ++tm) af[tm]  = load_frag(As, wm * 64 + tm * 16, lo, hi);
        #pragma unroll
        for (int tn = 0; tn < 4; ++tn) bfv[tn] = load_frag(Bs, wn * 64 + tn * 16, lo, hi);

        #pragma unroll
        for (int tm = 0; tm < 4; ++tm)
            #pragma unroll
            for (int tn = 0; tn < 4; ++tn)
                acc[tm][tn] = __builtin_amdgcn_wmma_f32_16x16x32_bf16(
                    false, af[tm], false, bfv[tn], (short)0, acc[tm][tn], false, false);
        __syncthreads();
    }

    // ---- epilogue: C layout VGPR i -> row = base + i + hi*8, col = base + lo ----
    if (mnfull) {
        #pragma unroll
        for (int tm = 0; tm < 4; ++tm) {
            #pragma unroll
            for (int tn = 0; tn < 4; ++tn) {
                const int col = n0 + wn * 64 + tn * 16 + lo;
                const float bv = bias ? bias[col] : 0.f;
                const int rb = m0 + wm * 64 + tm * 16 + hi * 8;
                #pragma unroll
                for (int i = 0; i < 8; ++i) {
                    float v = acc[tm][tn][i] + bv;
                    if (relu) v = fmaxf(v, 0.f);
                    Cb[(long)(rb + i) * ldC + col] = v;
                }
            }
        }
    } else if (!ALIGNED) {
        #pragma unroll
        for (int tm = 0; tm < 4; ++tm) {
            #pragma unroll
            for (int tn = 0; tn < 4; ++tn) {
                const int col = n0 + wn * 64 + tn * 16 + lo;
                if (col >= N) continue;
                const float bv = bias ? bias[col] : 0.f;
                const int rb = m0 + wm * 64 + tm * 16 + hi * 8;
                #pragma unroll
                for (int i = 0; i < 8; ++i) {
                    const int row = rb + i;
                    if (row < M) {
                        float v = acc[tm][tn][i] + bv;
                        if (relu) v = fmaxf(v, 0.f);
                        Cb[(long)row * ldC + col] = v;
                    }
                }
            }
        }
    }
}

// Row softmax over scores: v = v*scale (+causal mask), softmax along cols.
__global__ __launch_bounds__(256) void softmax_rows(
    float* __restrict__ Ptr, int cols, long rowStride, long batchStride,
    float scale, int causal)
{
    __shared__ float red[256];
    const int row = blockIdx.x;
    const long base = (long)blockIdx.y * batchStride + (long)row * rowStride;
    const int tid = threadIdx.x;

    float mx = -INFINITY;
    for (int c = tid; c < cols; c += 256) {
        float v = Ptr[base + c] * scale;
        if (causal && c > row) v = -INFINITY;
        Ptr[base + c] = v;
        mx = fmaxf(mx, v);
    }
    red[tid] = mx; __syncthreads();
    for (int s = 128; s > 0; s >>= 1) {
        if (tid < s) red[tid] = fmaxf(red[tid], red[tid + s]);
        __syncthreads();
    }
    mx = red[0]; __syncthreads();

    float sum = 0.f;
    for (int c = tid; c < cols; c += 256) {
        float e = __expf(Ptr[base + c] - mx);
        Ptr[base + c] = e;
        sum += e;
    }
    red[tid] = sum; __syncthreads();
    for (int s = 128; s > 0; s >>= 1) {
        if (tid < s) red[tid] += red[tid + s];
        __syncthreads();
    }
    const float inv = 1.f / red[0];
    for (int c = tid; c < cols; c += 256) Ptr[base + c] *= inv;
}

// x_out = LayerNorm(t + x_in) * g + b  over D_=768 (one block per row).
__global__ __launch_bounds__(256) void ln_residual(
    const float* __restrict__ t, const float* __restrict__ xin,
    const float* __restrict__ g, const float* __restrict__ b,
    float* __restrict__ xout)
{
    __shared__ float buf[D_];
    __shared__ float red[256];
    const long base = (long)blockIdx.x * D_;
    const int tid = threadIdx.x;

    float part = 0.f;
    for (int d = tid; d < D_; d += 256) {
        float v = t[base + d] + xin[base + d];
        buf[d] = v; part += v;
    }
    red[tid] = part; __syncthreads();
    for (int s = 128; s > 0; s >>= 1) { if (tid < s) red[tid] += red[tid + s]; __syncthreads(); }
    const float mu = red[0] / (float)D_; __syncthreads();

    part = 0.f;
    for (int d = tid; d < D_; d += 256) { float dv = buf[d] - mu; part += dv * dv; }
    red[tid] = part; __syncthreads();
    for (int s = 128; s > 0; s >>= 1) { if (tid < s) red[tid] += red[tid + s]; __syncthreads(); }
    const float rstd = rsqrtf(red[0] / (float)D_ + 1e-5f);

    for (int d = tid; d < D_; d += 256)
        xout[base + d] = g[d] * (buf[d] - mu) * rstd + b[d];
}

// x[b,s,:] = emb_table[tokens[b,s],:] + sinusoidal PE(s,:)
__global__ __launch_bounds__(256) void embed_kernel(
    const int* __restrict__ tokens, const float* __restrict__ table,
    float* __restrict__ x)
{
    const int row = blockIdx.x;            // b*S + s
    const int s = row % S_;
    const int tok = tokens[row];
    const long obase = (long)row * D_;
    const long tbase = (long)tok * D_;
    for (int d = threadIdx.x; d < D_; d += 256) {
        const float di = (float)(d & ~1);
        const float ang = (float)s * __powf(10000.f, -di / (float)D_);
        const float pe = (d & 1) ? __cosf(ang) : __sinf(ang);
        x[obase + d] = table[tbase + d] + pe;
    }
}

static inline void launch_gemm(hipStream_t st,
    const float* A, long ldA, long sAo, long sAi,
    const float* Bp, long ldbK, long ldbN, long sBo, long sBi,
    const float* bias, float* C, long ldC, long sCo, long sCi,
    int M, int N, int K, int batches, int inner, int relu)
{
    dim3 grid((unsigned)((N + 255) / 256), (unsigned)((M + 127) / 128), (unsigned)batches);
    const bool aligned = (M % 128 == 0) && (N % 256 == 0) && (K % 32 == 0);
    if (aligned)
        hipLaunchKernelGGL((wmma_gemm<true>), grid, dim3(256), 0, st,
                           A, ldA, sAo, sAi, Bp, ldbK, ldbN, sBo, sBi,
                           bias, C, ldC, sCo, sCi, M, N, K, inner, relu);
    else
        hipLaunchKernelGGL((wmma_gemm<false>), grid, dim3(256), 0, st,
                           A, ldA, sAo, sAi, Bp, ldbK, ldbN, sBo, sBi,
                           bias, C, ldC, sCo, sCi, M, N, K, inner, relu);
}

extern "C" void kernel_launch(void* const* d_in, const int* in_sizes, int n_in,
                              void* d_out, int out_size, void* d_ws, size_t ws_size,
                              hipStream_t stream) {
    (void)in_sizes; (void)n_in; (void)out_size; (void)ws_size;
    const int*   tokens = (const int*)  d_in[0];
    const float* img    = (const float*)d_in[1];
    const float* table  = (const float*)d_in[2];
    const float* qkvW   = (const float*)d_in[3];
    const float* qkvB   = (const float*)d_in[4];
    const float* spW    = (const float*)d_in[5];
    const float* spB    = (const float*)d_in[6];
    const float* slg    = (const float*)d_in[7];
    const float* slb    = (const float*)d_in[8];
    const float* mqW    = (const float*)d_in[9];
    const float* mqb    = (const float*)d_in[10];
    const float* mkW    = (const float*)d_in[11];
    const float* mkb    = (const float*)d_in[12];
    const float* mvW    = (const float*)d_in[13];
    const float* mvb    = (const float*)d_in[14];
    const float* cpW    = (const float*)d_in[15];
    const float* cpb    = (const float*)d_in[16];
    const float* clg    = (const float*)d_in[17];
    const float* clb    = (const float*)d_in[18];
    const float* f1W    = (const float*)d_in[19];
    const float* f1b    = (const float*)d_in[20];
    const float* f2W    = (const float*)d_in[21];
    const float* f2b    = (const float*)d_in[22];
    const float* pW     = (const float*)d_in[23];
    const float* pb     = (const float*)d_in[24];
    float* out = (float*)d_out;

    const int MR = B_ * S_;   // 4096 token rows
    const int MI = B_ * P_;   // 1568 image rows
    const float scale = 1.0f / sqrtf((float)D_);

    // workspace carve-up (bytes)
    char* ws = (char*)d_ws;
    size_t off = 0;
    auto carve = [&](size_t bytes) { float* p = (float*)(ws + off); off += (bytes + 255) & ~(size_t)255; return p; };
    float* X   = carve((size_t)MR * D_      * 4);  // activations [4096,768]
    float* QKV = carve((size_t)MR * 3 * D_  * 4);  // [4096,2304]
    float* SC  = carve((size_t)B_*H_*S_*S_  * 4);  // scores (reused for cross)
    float* O   = carve((size_t)MR * D_      * 4);  // attention output
    float* T   = carve((size_t)MR * D_      * 4);  // proj / q temp
    float* KI  = carve((size_t)MI * D_      * 4);  // img K
    float* VI  = carve((size_t)MI * D_      * 4);  // img V
    float* FFB = carve((size_t)MR * FF_     * 4);  // FFN mid

    // embedding + positional encoding
    hipLaunchKernelGGL(embed_kernel, dim3(MR), dim3(256), 0, stream, tokens, table, X);

    for (int l = 0; l < L_; ++l) {
        const float* qW  = qkvW + (size_t)l * D_ * 3 * D_;
        const float* qb  = qkvB + (size_t)l * 3 * D_;
        const float* sW  = spW  + (size_t)l * D_ * D_;
        const float* sb  = spB  + (size_t)l * D_;
        const float* g1  = slg  + (size_t)l * D_;
        const float* b1  = slb  + (size_t)l * D_;
        const float* mqw = mqW  + (size_t)l * D_ * D_;
        const float* mqbb= mqb  + (size_t)l * D_;
        const float* mkw = mkW  + (size_t)l * DI_ * D_;
        const float* mkbb= mkb  + (size_t)l * D_;
        const float* mvw = mvW  + (size_t)l * DI_ * D_;
        const float* mvbb= mvb  + (size_t)l * D_;
        const float* cW  = cpW  + (size_t)l * D_ * D_;
        const float* cb  = cpb  + (size_t)l * D_;
        const float* g2  = clg  + (size_t)l * D_;
        const float* b2  = clb  + (size_t)l * D_;
        const float* w1  = f1W  + (size_t)l * D_ * FF_;
        const float* bb1 = f1b  + (size_t)l * FF_;
        const float* w2  = f2W  + (size_t)l * FF_ * D_;
        const float* bb2 = f2b  + (size_t)l * D_;

        // ---- self attention ----
        launch_gemm(stream, X, D_, 0, 0, qW, 3 * D_, 1, 0, 0, qb,
                    QKV, 3 * D_, 0, 0, MR, 3 * D_, D_, 1, 1, 0);
        launch_gemm(stream,
                    QKV + 0,  3 * D_, (long)S_ * 3 * D_, HD_,
                    QKV + D_, 1, 3 * D_, (long)S_ * 3 * D_, HD_,
                    nullptr, SC, S_, (long)H_ * S_ * S_, (long)S_ * S_,
                    S_, S_, HD_, B_ * H_, H_, 0);
        hipLaunchKernelGGL(softmax_rows, dim3(S_, B_ * H_), dim3(256), 0, stream,
                           SC, S_, (long)S_, (long)S_ * S_, scale, 1);
        launch_gemm(stream,
                    SC, S_, (long)H_ * S_ * S_, (long)S_ * S_,
                    QKV + 2 * D_, 3 * D_, 1, (long)S_ * 3 * D_, HD_,
                    nullptr, O, HD_, (long)H_ * S_ * HD_, (long)S_ * HD_,
                    S_, HD_, S_, B_ * H_, H_, 0);
        launch_gemm(stream, O, D_, 0, 0, sW, D_, 1, 0, 0, sb,
                    T, D_, 0, 0, MR, D_, D_, 1, 1, 0);
        hipLaunchKernelGGL(ln_residual, dim3(MR), dim3(256), 0, stream, T, X, g1, b1, X);

        // ---- cross attention ----
        launch_gemm(stream, X, D_, 0, 0, mqw, D_, 1, 0, 0, mqbb,
                    T, D_, 0, 0, MR, D_, D_, 1, 1, 0);
        launch_gemm(stream, img, DI_, 0, 0, mkw, D_, 1, 0, 0, mkbb,
                    KI, D_, 0, 0, MI, D_, DI_, 1, 1, 0);
        launch_gemm(stream, img, DI_, 0, 0, mvw, D_, 1, 0, 0, mvbb,
                    VI, D_, 0, 0, MI, D_, DI_, 1, 1, 0);
        launch_gemm(stream,
                    T,  D_, (long)S_ * D_, HD_,
                    KI, 1, D_, (long)P_ * D_, HD_,
                    nullptr, SC, P_, (long)H_ * S_ * P_, (long)S_ * P_,
                    S_, P_, HD_, B_ * H_, H_, 0);
        hipLaunchKernelGGL(softmax_rows, dim3(S_, B_ * H_), dim3(256), 0, stream,
                           SC, P_, (long)P_, (long)S_ * P_, scale, 0);
        launch_gemm(stream,
                    SC, P_, (long)H_ * S_ * P_, (long)S_ * P_,
                    VI, D_, 1, (long)P_ * D_, HD_,
                    nullptr, O, HD_, (long)H_ * S_ * HD_, (long)S_ * HD_,
                    S_, HD_, P_, B_ * H_, H_, 0);
        launch_gemm(stream, O, D_, 0, 0, cW, D_, 1, 0, 0, cb,
                    T, D_, 0, 0, MR, D_, D_, 1, 1, 0);
        hipLaunchKernelGGL(ln_residual, dim3(MR), dim3(256), 0, stream, T, X, g2, b2, X);

        // ---- feed forward (no residual) ----
        launch_gemm(stream, X, D_, 0, 0, w1, FF_, 1, 0, 0, bb1,
                    FFB, FF_, 0, 0, MR, FF_, D_, 1, 1, 1);
        launch_gemm(stream, FFB, FF_, 0, 0, w2, D_, 1, 0, 0, bb2,
                    X, D_, 0, 0, MR, D_, FF_, 1, 1, 0);
    }

    // final vocab projection: out = X @ proj_W + proj_b : [4096, 32000]
    launch_gemm(stream, X, D_, 0, 0, pW, V_, 1, 0, 0, pb,
                out, V_, 0, 0, MR, V_, D_, 1, 1, 0);
}
